// HGNNModel_2layer_53352083751414
// MI455X (gfx1250) — matrix operand order repaired and checked by reading, not compiled, not run on hardware
//
#include <hip/hip_runtime.h>
#include <hip/hip_bf16.h>

// ---------------- problem constants ----------------
#define NN 8192      // vertices
#define EE 4096      // hyperedges
#define VH 128       // input feature dim
#define H0 64        // hidden dim
#define BN_EPS 1e-5f

typedef __attribute__((ext_vector_type(16))) __bf16 v16bf;
typedef __attribute__((ext_vector_type(8)))  float  v8f;

// ---------------- CDNA5 async global->LDS path (ASYNCcnt), with safe fallback ----------------
#if defined(__has_builtin)
#if __has_builtin(__builtin_amdgcn_global_load_async_to_lds_b128) && \
    __has_builtin(__builtin_amdgcn_s_wait_asynccnt)
#define USE_ASYNC_LDS 1
#endif
#endif

#if defined(USE_ASYNC_LDS)
typedef int b128v __attribute__((vector_size(16)));          // 16-byte payload type
typedef __attribute__((address_space(1))) b128v* gptr128;    // global
typedef __attribute__((address_space(3))) b128v* lptr128;    // LDS
#endif

// copy one 16-byte chunk (8 bf16) global -> LDS
__device__ __forceinline__ void async_copy16(const __bf16* __restrict__ g, __bf16* l) {
#if defined(USE_ASYNC_LDS)
  __builtin_amdgcn_global_load_async_to_lds_b128(
      (gptr128)(void*)g, (lptr128)(void*)l, 0, 0);
#else
#pragma unroll
  for (int q = 0; q < 8; ++q) l[q] = g[q];
#endif
}

__device__ __forceinline__ void wait_async_lds() {
#if defined(USE_ASYNC_LDS)
  __builtin_amdgcn_s_wait_asynccnt(0);
#endif
}

// ---------------- WMMA fragment helpers (wave32; A and B share the contiguous k = 16h+s order) ----------------
// A-matrix 16x32 bf16 (row-major src, stride ld). Lane l: m = l&15, h = l>>4.
// elems 0..7  -> K = 16h+0..7   (VGPR0..3)
// elems 8..15 -> K = 16h+8..15 (VGPR4..7)
__device__ __forceinline__ v16bf load_a_bf16(const __bf16* __restrict__ base, int ld, int lane) {
  const int m = lane & 15, h = lane >> 4;
  const __bf16* r = base + (size_t)m * ld + 16 * h;
  v16bf a;
#pragma unroll
  for (int i = 0; i < 8; ++i) a[i] = r[i];
#pragma unroll
  for (int i = 0; i < 8; ++i) a[8 + i] = r[8 + i];
  return a;
}

__device__ __forceinline__ v16bf load_a_f32(const float* __restrict__ base, int ld, int lane) {
  const int m = lane & 15, h = lane >> 4;
  const float* r = base + (size_t)m * ld + 16 * h;
  v16bf a;
#pragma unroll
  for (int i = 0; i < 8; ++i) a[i] = (__bf16)r[i];
#pragma unroll
  for (int i = 0; i < 8; ++i) a[8 + i] = (__bf16)r[8 + i];
  return a;
}

// B-matrix 32x16 bf16 (row-major src, stride ld). Lane l: n = l&15, h = l>>4.
// VGPR v holds K = 16h+2v, 16h+2v+1 for column n.
__device__ __forceinline__ v16bf load_b_bf16(const __bf16* __restrict__ base, int ld, int lane) {
  const int n = lane & 15, h = lane >> 4;
  v16bf b;
#pragma unroll
  for (int v = 0; v < 8; ++v) {
    b[2 * v]     = base[(size_t)(16 * h + 2 * v) * ld + n];
    b[2 * v + 1] = base[(size_t)(16 * h + 2 * v + 1) * ld + n];
  }
  return b;
}

__device__ __forceinline__ v16bf load_b_f32(const float* __restrict__ base, int ld, int lane) {
  const int n = lane & 15, h = lane >> 4;
  v16bf b;
#pragma unroll
  for (int v = 0; v < 8; ++v) {
    b[2 * v]     = (__bf16)base[(size_t)(16 * h + 2 * v) * ld + n];
    b[2 * v + 1] = (__bf16)base[(size_t)(16 * h + 2 * v + 1) * ld + n];
  }
  return b;
}

#define WMMA_BF16(a, b, c) \
  __builtin_amdgcn_wmma_f32_16x16x32_bf16(false, (a), false, (b), (short)0, (c), false, false)

// ---------------- kernel 0: H (f32 0/1, 128 MB) -> bf16 (64 MB, exact, L2-resident) ----------------
__global__ void k_cvt_bf16(const float* __restrict__ src, __bf16* __restrict__ dst) {
  const size_t i = ((size_t)blockIdx.x * blockDim.x + threadIdx.x) * 8;
#pragma unroll
  for (int k = 0; k < 8; ++k) dst[i + k] = (__bf16)src[i + k];
}

// ---------------- kernel 1: Pt1[j,i] = bf16(Dv[i] * (X@theta1)[i,j]), stored transposed [H0, N] ----
__global__ void k_gemm1(const float* __restrict__ X, const float* __restrict__ th1,
                        const float* __restrict__ Dv, __bf16* __restrict__ Pt1) {
  const int lane = threadIdx.x & 31, wave = threadIdx.x >> 5;
  const int i0 = blockIdx.x * 64 + wave * 16;
  v8f acc[4] = {};
#pragma unroll
  for (int kk = 0; kk < VH; kk += 32) {
    v16bf a = load_a_f32(X + (size_t)i0 * VH + kk, VH, lane);
#pragma unroll
    for (int t = 0; t < 4; ++t) {
      v16bf b = load_b_f32(th1 + (size_t)kk * H0 + t * 16, H0, lane);
      acc[t] = WMMA_BF16(a, b, acc[t]);
    }
  }
  const int h = lane >> 4, nn = lane & 15;
#pragma unroll
  for (int t = 0; t < 4; ++t)
#pragma unroll
    for (int r = 0; r < 8; ++r) {
      const int i = i0 + 8 * h + r;
      const int j = t * 16 + nn;
      Pt1[(size_t)j * NN + i] = (__bf16)(Dv[i] * acc[t][r]);
    }
}

// ---------------- kernel 2: Mt[MR, E] = Pt[MR, N] @ Hbf[N, E]  (== (H^T P)^T) ------------------
// block = 128 threads (4 waves); block owns 64 e-columns; wave owns one 16-wide n-subtile;
// Hbf tile [32 x 64] double-buffered in LDS via async global->LDS copies.
__device__ __forceinline__ void stageA(__bf16* dst, const __bf16* __restrict__ Hb,
                                       int k0, int n0) {
  // Hb[k0..k0+32, n0..n0+64] -> dst[32*64]; 256 x 16B chunks over 128 threads
#pragma unroll
  for (int q = 0; q < 2; ++q) {
    const int c = threadIdx.x + q * 128;
    const int row = c >> 3, ch = c & 7;
    async_copy16(Hb + (size_t)(k0 + row) * EE + n0 + ch * 8, dst + row * 64 + ch * 8);
  }
}

template <int MR>
__global__ void k_gemmA(const __bf16* __restrict__ Pt, const __bf16* __restrict__ Hb,
                        float* __restrict__ Mt) {
  __shared__ __align__(16) __bf16 lds[2][32 * 64];
  const int lane = threadIdx.x & 31, wave = threadIdx.x >> 5;
  const int n0 = blockIdx.x * 64;
  const int nw = wave * 16;
  v8f acc[MR / 16] = {};

  stageA(lds[0], Hb, 0, n0);
  int buf = 0;
  for (int k0 = 0; k0 < NN; k0 += 32) {
    wait_async_lds();
    __syncthreads();
    if (k0 + 32 < NN) stageA(lds[buf ^ 1], Hb, k0 + 32, n0);
    v16bf b = load_b_bf16(lds[buf] + nw, 64, lane);
#pragma unroll
    for (int t = 0; t < MR / 16; ++t) {
      v16bf a = load_a_bf16(Pt + (size_t)(t * 16) * NN + k0, NN, lane);
      acc[t] = WMMA_BF16(a, b, acc[t]);
    }
    buf ^= 1;
  }
  const int h = lane >> 4, nn = lane & 15;
#pragma unroll
  for (int t = 0; t < MR / 16; ++t)
#pragma unroll
    for (int r = 0; r < 8; ++r) {
      const int j = t * 16 + 8 * h + r;
      Mt[(size_t)j * EE + n0 + nw + nn] = acc[t][r];
    }
}

// ---------------- kernel 3: Mb[e, j] = bf16(Mt[j, e] * W[e] * De_inv[e])  [E, MR] row-major ------
template <int MR>
__global__ void k_scaleT(const float* __restrict__ Mt, const float* __restrict__ W,
                         const float* __restrict__ De, __bf16* __restrict__ Mb) {
  const int idx = blockIdx.x * blockDim.x + threadIdx.x;
  const int e = idx / MR, j = idx % MR;
  Mb[idx] = (__bf16)(Mt[(size_t)j * EE + e] * W[e] * De[e]);
}

// ---------------- kernel 4: out[i, j] = epi(Dv[i] * (Hbf @ Mb)[i, j])  [N, MR] -------------------
// block = 256 threads (8 waves); wave owns 16 output rows; Mb tile [32 x MR] double-buffered in LDS.
template <int MR>
__device__ __forceinline__ void stageB(__bf16* dst, const __bf16* __restrict__ Mb, int k0) {
  // contiguous 32*MR bf16 -> dst; 16B chunks over 256 threads
  for (int c = threadIdx.x; c < (32 * MR) / 8; c += 256)
    async_copy16(Mb + (size_t)k0 * MR + c * 8, dst + c * 8);
}

template <int MR, int MODE>
__global__ void k_gemmB(const __bf16* __restrict__ Hb, const __bf16* __restrict__ Mb,
                        const float* __restrict__ Dv, float* __restrict__ out) {
  __shared__ __align__(16) __bf16 lds[2][32 * MR];
  const int lane = threadIdx.x & 31, wave = threadIdx.x >> 5;
  const int row0 = blockIdx.x * 128 + wave * 16;
  v8f acc[MR / 16] = {};

  stageB<MR>(lds[0], Mb, 0);
  int buf = 0;
  for (int k0 = 0; k0 < EE; k0 += 32) {
    wait_async_lds();
    __syncthreads();
    if (k0 + 32 < EE) stageB<MR>(lds[buf ^ 1], Mb, k0 + 32);
    v16bf a = load_a_bf16(Hb + (size_t)row0 * EE + k0, EE, lane);
#pragma unroll
    for (int t = 0; t < MR / 16; ++t) {
      v16bf b = load_b_bf16(lds[buf] + t * 16, MR, lane);
      acc[t] = WMMA_BF16(a, b, acc[t]);
    }
    buf ^= 1;
  }
  const int h = lane >> 4, nn = lane & 15;
#pragma unroll
  for (int t = 0; t < MR / 16; ++t)
#pragma unroll
    for (int r = 0; r < 8; ++r) {
      const int row = row0 + 8 * h + r;
      float v = Dv[row] * acc[t][r];
      if (MODE == 0) v = (v > 0.f) ? v : 0.01f * v;
      out[(size_t)row * MR + t * 16 + nn] = v;
    }
}

// ---------------- kernel 5: per-column BN stats over X1 [N, H0] ----------------------------------
__global__ void k_bnstats(const float* __restrict__ X1, float* __restrict__ mean,
                          float* __restrict__ rstd) {
  __shared__ float s1[256], s2[256];
  const int j = blockIdx.x;
  float a = 0.f, b = 0.f;
  for (int i = threadIdx.x; i < NN; i += 256) {
    const float x = X1[(size_t)i * H0 + j];
    a += x; b += x * x;
  }
  s1[threadIdx.x] = a; s2[threadIdx.x] = b;
  __syncthreads();
  for (int s = 128; s > 0; s >>= 1) {
    if (threadIdx.x < s) { s1[threadIdx.x] += s1[threadIdx.x + s]; s2[threadIdx.x] += s2[threadIdx.x + s]; }
    __syncthreads();
  }
  if (threadIdx.x == 0) {
    const float mu = s1[0] / NN;
    const float var = s2[0] / NN - mu * mu;
    mean[j] = mu;
    rstd[j] = rsqrtf(var + BN_EPS);
  }
}

// ---------------- kernel 6: BN + dot(theta2) + Dv scale -> padded Pt2 [16, N] bf16 ---------------
__global__ void k_vec2(const float* __restrict__ X1, const float* __restrict__ mean,
                       const float* __restrict__ rstd, const float* __restrict__ gamma,
                       const float* __restrict__ beta, const float* __restrict__ th2,
                       const float* __restrict__ Dv, __bf16* __restrict__ Pt2) {
  const int i = blockIdx.x * blockDim.x + threadIdx.x;
  float acc = 0.f;
#pragma unroll
  for (int j = 0; j < H0; ++j) {
    float x = X1[(size_t)i * H0 + j];
    x = (x - mean[j]) * rstd[j] * gamma[j] + beta[j];
    acc += x * th2[j];
  }
  Pt2[i] = (__bf16)(Dv[i] * acc);
#pragma unroll
  for (int r = 1; r < 16; ++r) Pt2[(size_t)r * NN + i] = (__bf16)0.f;
}

// ---------------- kernel 7: out[i] = sigmoid(Y2[i, 0]) -------------------------------------------
__global__ void k_sigmoid(const float* __restrict__ Y2, float* __restrict__ out) {
  const int i = blockIdx.x * blockDim.x + threadIdx.x;
  out[i] = 1.f / (1.f + __expf(-Y2[(size_t)i * 16]));
}

// ---------------- host launch --------------------------------------------------------------------
extern "C" void kernel_launch(void* const* d_in, const int* in_sizes, int n_in,
                              void* d_out, int out_size, void* d_ws, size_t ws_size,
                              hipStream_t stream) {
  const float* X   = (const float*)d_in[0];   // [N, VH]
  const float* Dv  = (const float*)d_in[1];   // [N]
  const float* De  = (const float*)d_in[2];   // [E]
  const float* H   = (const float*)d_in[3];   // [N, E]
  const float* W   = (const float*)d_in[4];   // [E]
  const float* th1 = (const float*)d_in[5];   // [VH, H0]
  const float* th2 = (const float*)d_in[6];   // [H0, 1]
  const float* gam = (const float*)d_in[7];   // [H0]
  const float* bet = (const float*)d_in[8];   // [H0]
  float* out = (float*)d_out;                 // [N]

  char* ws = (char*)d_ws;
  size_t off = 0;
  __bf16* Hbf = (__bf16*)(ws + off); off += (size_t)NN * EE * 2;      // 64 MB (L2-resident)
  __bf16* Pt1 = (__bf16*)(ws + off); off += (size_t)H0 * NN * 2;      // 1 MB
  float*  Mt1 = (float*) (ws + off); off += (size_t)H0 * EE * 4;      // 1 MB
  __bf16* Mb1 = (__bf16*)(ws + off); off += (size_t)EE * H0 * 2;      // 0.5 MB
  float*  X1  = (float*) (ws + off); off += (size_t)NN * H0 * 4;      // 2 MB
  float*  mu  = (float*) (ws + off); off += 256;
  float*  rs  = (float*) (ws + off); off += 256;
  __bf16* Pt2 = (__bf16*)(ws + off); off += (size_t)16 * NN * 2;      // 256 KB
  float*  Mt2 = (float*) (ws + off); off += (size_t)16 * EE * 4;      // 256 KB
  __bf16* Mb2 = (__bf16*)(ws + off); off += (size_t)EE * 16 * 2;      // 128 KB
  float*  Y2  = (float*) (ws + off); off += (size_t)NN * 16 * 4;      // 512 KB

  // H -> bf16 (exact: entries are 0/1); subsequent GEMM passes stream it from L2
  k_cvt_bf16<<<((size_t)NN * EE / 8) / 256, 256, 0, stream>>>(H, Hbf);

  // ---- layer 1 ----
  k_gemm1<<<NN / 64, 128, 0, stream>>>(X, th1, Dv, Pt1);                 // Pt1 = (Dv*(X@th1))^T
  k_gemmA<H0><<<EE / 64, 128, 0, stream>>>(Pt1, Hbf, Mt1);               // Mt1 = Pt1 @ Hbf
  k_scaleT<H0><<<(EE * H0) / 256, 256, 0, stream>>>(Mt1, W, De, Mb1);    // Mb1 = (W*De) * Mt1^T
  k_gemmB<H0, 0><<<NN / 128, 256, 0, stream>>>(Hbf, Mb1, Dv, X1);        // X1 = leaky(Dv * Hbf@Mb1)
  k_bnstats<<<H0, 256, 0, stream>>>(X1, mu, rs);

  // ---- layer 2 (h=1 padded to 16-wide WMMA tile) ----
  k_vec2<<<NN / 256, 256, 0, stream>>>(X1, mu, rs, gam, bet, th2, Dv, Pt2);
  k_gemmA<16><<<EE / 64, 128, 0, stream>>>(Pt2, Hbf, Mt2);
  k_scaleT<16><<<(EE * 16) / 256, 256, 0, stream>>>(Mt2, W, De, Mb2);
  k_gemmB<16, 1><<<NN / 128, 256, 0, stream>>>(Hbf, Mb2, Dv, Y2);
  k_sigmoid<<<NN / 256, 256, 0, stream>>>(Y2, out);
}
